// NodeEmbedding_69380901699623
// MI455X (gfx1250) — matrix operand order repaired and checked by reading, compiled, not run on hardware
//
#include <hip/hip_runtime.h>
#include <hip/hip_bf16.h>
#include <cmath>

// ---------------------------------------------------------------------------
// CDNA5 (gfx1250) wave32 WMMA implementation of the NodeEmbedding forward.
// All GEMMs run through v_wmma_f32_16x16x32_bf16 (bf16 in, f32 accumulate).
// Every GEMM B operand is stored transposed [N,K] row-major so ALL fragment
// loads are contiguous global_load_b128 (no strided u16 B loads).
// ---------------------------------------------------------------------------

typedef __bf16 bf16_t;
typedef __attribute__((ext_vector_type(16))) __bf16 v16bf;
typedef __attribute__((ext_vector_type(8)))  float  v8f;

#define N_NODES 8192
#define EMB     256
#define DIN     384
#define NHEAD   4
#define DH      128      // per-head dim == WGAT output dim
#define HD      512      // NHEAD * DH
#define NEDGE   131072
#define CHUNK   32768
#define QBLK    256      // query rows per attention chunk

// ---------------------------------------------------------------------------
// WMMA GEMM:  C[M,N] = A[M,K] * Bt[N,K]^T + bias
// Block = 128 threads = 4 waves arranged 2x2; each wave owns a 16x32 C strip
// (two 16x16 WMMA accumulators sharing one A fragment).  Block tile 32x64.
// M multiple of 32, N multiple of 64, K multiple of 32.
// ---------------------------------------------------------------------------
__global__ void __launch_bounds__(128)
k_gemm(const bf16_t* __restrict__ A, const bf16_t* __restrict__ Bt,
       float* __restrict__ C, int M, int N, int K,
       int lda, int ldb, int ldc, const float* __restrict__ bias)
{
    const int lane = threadIdx.x & 31;
    const int wave = threadIdx.x >> 5;
    const int m0 = (blockIdx.y * 2 + (wave >> 1)) * 16;
    const int n0 = (blockIdx.x * 2 + (wave & 1)) * 32;
    if (m0 >= M || n0 >= N) return;

    const int hw = lane >> 4;    // half-wave: 0 or 1
    const int lr = lane & 15;

    v8f acc0 = {}, acc1 = {};
    for (int k0 = 0; k0 < K; k0 += 32) {
        // A fragment (16x32, MxK): lane holds row m0+lr,
        // K elements k0 + 16*hw + {0..15} (same k order as B)
        const bf16_t* ap = A + (size_t)(m0 + lr) * lda + k0 + hw * 16;
        __builtin_prefetch(ap + 32, 0, 1);   // global_prefetch next K tile
        v16bf a;
#pragma unroll
        for (int i = 0; i < 8; ++i) a[i] = ap[i];
#pragma unroll
        for (int i = 0; i < 8; ++i) a[8 + i] = ap[8 + i];

        // B fragments (32x16, KxN): lane holds col n0+lr (resp. +16),
        // K elements k0 + 16*hw + {0..15}  -> 32 contiguous bytes per lane
        const bf16_t* bp0 = Bt + (size_t)(n0 + lr) * ldb + k0 + hw * 16;
        const bf16_t* bp1 = bp0 + (size_t)16 * ldb;
        __builtin_prefetch(bp0 + 32, 0, 1);
        v16bf b0, b1;
#pragma unroll
        for (int i = 0; i < 16; ++i) b0[i] = bp0[i];
#pragma unroll
        for (int i = 0; i < 16; ++i) b1[i] = bp1[i];

        acc0 = __builtin_amdgcn_wmma_f32_16x16x32_bf16(
                   false, a, false, b0, (short)0, acc0, false, false);
        acc1 = __builtin_amdgcn_wmma_f32_16x16x32_bf16(
                   false, a, false, b1, (short)0, acc1, false, false);
    }

    // C fragment: lane -> N = n0+lr, VGPR r -> M = m0 + r + 8*hw
    const int mb = m0 + hw * 8;
    const int na = n0 + lr;
    const int nb = n0 + 16 + lr;
    const float bva = bias ? bias[na] : 0.0f;
    const float bvb = bias ? bias[nb] : 0.0f;
#pragma unroll
    for (int r = 0; r < 8; ++r) {
        C[(size_t)(mb + r) * ldc + na] = acc0[r] + bva;
        C[(size_t)(mb + r) * ldc + nb] = acc1[r] + bvb;
    }
}

// ---------------------------------------------------------------------------
// z = [tile(x,128) | embed[ids]]  -> bf16 [N, 384]
// ---------------------------------------------------------------------------
__global__ void __launch_bounds__(DIN)
k_prep(const int* __restrict__ ids, const float* __restrict__ x,
       const float* __restrict__ embed, bf16_t* __restrict__ z)
{
    const int n = blockIdx.x, c = threadIdx.x;
    float v = (c < DH) ? x[n] : embed[(size_t)ids[n] * EMB + (c - DH)];
    z[(size_t)n * DIN + c] = (bf16_t)v;
}

// split qkv[N,1152]: q bf16 row-major (pre-scaled), k bf16 row-major,
// v written TRANSPOSED ([DIN, N_NODES]) so the P*V GEMM gets contiguous B.
__global__ void __launch_bounds__(DIN)
k_split_qkv(const float* __restrict__ qkv, bf16_t* __restrict__ q,
            bf16_t* __restrict__ k, bf16_t* __restrict__ vT, float qscale)
{
    const int n = blockIdx.x, c = threadIdx.x;
    const float* row = qkv + (size_t)n * (3 * DIN);
    q[(size_t)n * DIN + c]       = (bf16_t)(row[c] * qscale);
    k[(size_t)n * DIN + c]       = (bf16_t)(row[DIN + c]);
    vT[(size_t)c * N_NODES + n]  = (bf16_t)(row[2 * DIN + c]);
}

// f32 -> bf16 element copy
__global__ void k_cvt(const float* __restrict__ s, bf16_t* __restrict__ d, long n)
{
    long i = (long)blockIdx.x * blockDim.x + threadIdx.x;
    if (i < n) d[i] = (bf16_t)s[i];
}

// f32 [R,C] row-major -> bf16 transposed [C,R]
__global__ void k_cvt_t(const float* __restrict__ s, bf16_t* __restrict__ d,
                        int R, int Cc)
{
    int idx = blockIdx.x * blockDim.x + threadIdx.x;
    if (idx >= R * Cc) return;
    const int r = idx / Cc, c = idx - r * Cc;
    d[(size_t)c * R + r] = (bf16_t)s[idx];
}

__global__ void k_fill0(float* __restrict__ p, long n)
{
    for (long i = (long)blockIdx.x * blockDim.x + threadIdx.x; i < n;
         i += (long)gridDim.x * blockDim.x) p[i] = 0.0f;
}

__global__ void k_copyf(const float* __restrict__ s, float* __restrict__ d, long n)
{
    for (long i = (long)blockIdx.x * blockDim.x + threadIdx.x; i < n;
         i += (long)gridDim.x * blockDim.x) d[i] = s[i];
}

__global__ void k_i2f(const int* __restrict__ s, float* __restrict__ d, int n)
{
    int i = blockIdx.x * blockDim.x + threadIdx.x;
    if (i < n) d[i] = (float)s[i];
}

// ---------------------------------------------------------------------------
// row softmax over [rows, cols] f32 -> bf16 (cols = 8192), one block per row
// ---------------------------------------------------------------------------
__global__ void __launch_bounds__(256)
k_row_softmax(const float* __restrict__ S, bf16_t* __restrict__ P, int cols)
{
    __shared__ float sh[256];
    const int row = blockIdx.x, t = threadIdx.x;
    const float* s = S + (size_t)row * cols;

    float m = -3.4e38f;
    for (int c = t; c < cols; c += 256) m = fmaxf(m, s[c]);
    sh[t] = m; __syncthreads();
    for (int st = 128; st > 0; st >>= 1) {
        if (t < st) sh[t] = fmaxf(sh[t], sh[t + st]); __syncthreads();
    }
    m = sh[0]; __syncthreads();

    float sum = 0.0f;
    for (int c = t; c < cols; c += 256) sum += __expf(s[c] - m);
    sh[t] = sum; __syncthreads();
    for (int st = 128; st > 0; st >>= 1) {
        if (t < st) sh[t] += sh[t + st]; __syncthreads();
    }
    const float inv = 1.0f / sh[0];

    bf16_t* p = P + (size_t)row * cols;
    for (int c = t; c < cols; c += 256) p[c] = (bf16_t)(__expf(s[c] - m) * inv);
}

// ---------------------------------------------------------------------------
// emb = silu(LN(fused)) + embed[ids]  (row width 256), writes f32 + bf16
// ---------------------------------------------------------------------------
__global__ void __launch_bounds__(EMB)
k_ln_silu_add(const float* __restrict__ fused, const float* __restrict__ g,
              const float* __restrict__ bta, const float* __restrict__ embed,
              const int* __restrict__ ids, float* __restrict__ ef,
              bf16_t* __restrict__ eb)
{
    __shared__ float sh[EMB];
    const int row = blockIdx.x, c = threadIdx.x;
    const float v = fused[(size_t)row * EMB + c];

    sh[c] = v; __syncthreads();
    for (int st = EMB / 2; st > 0; st >>= 1) {
        if (c < st) sh[c] += sh[c + st]; __syncthreads();
    }
    const float mean = sh[0] / EMB; __syncthreads();

    const float d = v - mean;
    sh[c] = d * d; __syncthreads();
    for (int st = EMB / 2; st > 0; st >>= 1) {
        if (c < st) sh[c] += sh[c + st]; __syncthreads();
    }
    const float var = sh[0] / EMB;

    float y = d * rsqrtf(var + 1e-5f) * g[c] + bta[c];
    y = y / (1.0f + __expf(-y));                       // SiLU
    const float e = y + embed[(size_t)ids[row] * EMB + c];
    ef[(size_t)row * EMB + c] = e;
    eb[(size_t)row * EMB + c] = (bf16_t)e;
}

// final LN over (out2 + 0.5*(s0+s1)) rows of width 128 -> d_out
__global__ void __launch_bounds__(DH)
k_final_ln(const float* __restrict__ o2, const float* __restrict__ s0,
           const float* __restrict__ s1, const float* __restrict__ g,
           const float* __restrict__ b, float* __restrict__ dst)
{
    __shared__ float sh[DH];
    const int row = blockIdx.x, c = threadIdx.x;
    const size_t i = (size_t)row * DH + c;
    const float v = o2[i] + 0.5f * (s0[i] + s1[i]);

    sh[c] = v; __syncthreads();
    for (int st = DH / 2; st > 0; st >>= 1) {
        if (c < st) sh[c] += sh[c + st]; __syncthreads();
    }
    const float mean = sh[0] / DH; __syncthreads();

    const float d = v - mean;
    sh[c] = d * d; __syncthreads();
    for (int st = DH / 2; st > 0; st >>= 1) {
        if (c < st) sh[c] += sh[c + st]; __syncthreads();
    }
    const float var = sh[0] / DH;
    dst[i] = d * rsqrtf(var + 1e-5f) * g[c] + b[c];
}

// ---------------------------------------------------------------------------
// WGAT helpers: fold attn into W  (a = (x@W) dot attn_h  ==  x dot (W attn_h))
// ---------------------------------------------------------------------------
__global__ void k_wfold(const float* __restrict__ wmat, const float* __restrict__ bvec,
                        const float* __restrict__ attn, float* __restrict__ wf,
                        float* __restrict__ bfold, int in_dim)
{
    int idx = blockIdx.x * blockDim.x + threadIdx.x;
    if (idx >= (in_dim + 1) * NHEAD) return;
    const int c = idx >> 2, h = idx & 3;
    const float* av = attn + h * DH;
    float s = 0.0f;
    if (c < in_dim) {
        const float* wr = wmat + (size_t)c * HD + h * DH;
        for (int d = 0; d < DH; ++d) s += wr[d] * av[d];
        wf[c * NHEAD + h] = s;
    } else {
        const float* br = bvec + h * DH;
        for (int d = 0; d < DH; ++d) s += br[d] * av[d];
        bfold[h] = s;
    }
}

__global__ void __launch_bounds__(256)
k_alpha(const float* __restrict__ xin, const float* __restrict__ wf,
        const float* __restrict__ bfold, float* __restrict__ alpha, int in_dim)
{
    int idx = blockIdx.x * 256 + threadIdx.x;       // n*4 + h
    if (idx >= N_NODES * NHEAD) return;
    const int n = idx >> 2, h = idx & 3;
    const float* xr = xin + (size_t)n * in_dim;
    float s = bfold[h];
    for (int c = 0; c < in_dim; ++c) s += xr[c] * wf[c * NHEAD + h];
    alpha[idx] = s;
}

__global__ void k_edge_logits(const float* __restrict__ ai, const float* __restrict__ aj,
                              const int* __restrict__ src, const int* __restrict__ dst,
                              const float* __restrict__ ea, float* __restrict__ a)
{
    int e = blockIdx.x * blockDim.x + threadIdx.x;
    if (e >= NEDGE) return;
    const int s = src[e], d = dst[e];
    const float w = ea[e];
#pragma unroll
    for (int h = 0; h < NHEAD; ++h)
        a[(size_t)e * NHEAD + h] = (ai[d * NHEAD + h] + aj[s * NHEAD + h]) * w;
}

// softmax over CHUNK edges per (chunk, head): block = (chunk*4 + h)
__global__ void __launch_bounds__(256)
k_chunk_softmax(float* __restrict__ a)
{
    __shared__ float sh[256];
    const int chunk = blockIdx.x >> 2, h = blockIdx.x & 3, t = threadIdx.x;
    float* base = a + (size_t)chunk * CHUNK * NHEAD + h;

    float m = -3.4e38f;
    for (int i = t; i < CHUNK; i += 256) m = fmaxf(m, base[(size_t)i * NHEAD]);
    sh[t] = m; __syncthreads();
    for (int st = 128; st > 0; st >>= 1) {
        if (t < st) sh[t] = fmaxf(sh[t], sh[t + st]); __syncthreads();
    }
    m = sh[0]; __syncthreads();

    float sum = 0.0f;
    for (int i = t; i < CHUNK; i += 256) sum += __expf(base[(size_t)i * NHEAD] - m);
    sh[t] = sum; __syncthreads();
    for (int st = 128; st > 0; st >>= 1) {
        if (t < st) sh[t] += sh[t + st]; __syncthreads();
    }
    const float inv = 1.0f / sh[0];
    for (int i = t; i < CHUNK; i += 256) {
        const size_t j = (size_t)i * NHEAD;
        base[j] = __expf(base[j] - m) * inv;
    }
}

// msg[e,d] = sum_h t[src[e], h*128+d] * a[e,h]; atomic add into out[dst[e], d]
__global__ void __launch_bounds__(DH)
k_scatter(const float* __restrict__ t, const float* __restrict__ a,
          const int* __restrict__ src, const int* __restrict__ dst,
          float* __restrict__ out)
{
    const int e = blockIdx.x, d = threadIdx.x;
    const int s = src[e], dn = dst[e];
    const float* tr = t + (size_t)s * HD;
    const float* ae = a + (size_t)e * NHEAD;
    const float m = tr[d]          * ae[0] + tr[DH + d]     * ae[1] +
                    tr[2 * DH + d] * ae[2] + tr[3 * DH + d] * ae[3];
    atomicAdd(out + (size_t)dn * DH + d, m);
}

// ---------------------------------------------------------------------------
// Host orchestration
// ---------------------------------------------------------------------------
extern "C" void kernel_launch(void* const* d_in, const int* in_sizes, int n_in,
                              void* d_out, int out_size, void* d_ws, size_t ws_size,
                              hipStream_t stream)
{
    (void)in_sizes; (void)n_in; (void)out_size; (void)ws_size;

    const int*   ids       = (const int*)  d_in[0];
    const float* x         = (const float*)d_in[1];
    const float* edge_attr = (const float*)d_in[2];
    const int*   edge_idx  = (const int*)  d_in[3];
    const float* embed     = (const float*)d_in[4];
    const float* qkv_w     = (const float*)d_in[5];
    const float* qkv_b     = (const float*)d_in[6];
    const float* proj_w    = (const float*)d_in[7];
    const float* proj_b    = (const float*)d_in[8];
    const float* fuse_w    = (const float*)d_in[9];
    const float* fuse_b    = (const float*)d_in[10];
    const float* ln1_g     = (const float*)d_in[11];
    const float* ln1_b     = (const float*)d_in[12];
    const float* wi0 = (const float*)d_in[13]; const float* bi0 = (const float*)d_in[14];
    const float* wj0 = (const float*)d_in[15]; const float* bj0 = (const float*)d_in[16];
    const float* wt0 = (const float*)d_in[17]; const float* bt0 = (const float*)d_in[18];
    const float* attn0 = (const float*)d_in[19];
    const float* skip0_w = (const float*)d_in[20]; const float* skip0_b = (const float*)d_in[21];
    const float* wi1 = (const float*)d_in[22]; const float* bi1 = (const float*)d_in[23];
    const float* wj1 = (const float*)d_in[24]; const float* bj1 = (const float*)d_in[25];
    const float* wt1 = (const float*)d_in[26]; const float* bt1 = (const float*)d_in[27];
    const float* attn1 = (const float*)d_in[28];
    const float* skip1_w = (const float*)d_in[29]; const float* skip1_b = (const float*)d_in[30];
    const float* ln2_g = (const float*)d_in[31]; const float* ln2_b = (const float*)d_in[32];

    const int* src = edge_idx;           // edge_index[0]
    const int* dst = edge_idx + NEDGE;   // edge_index[1]

    // ---- workspace carving ----
    unsigned char* wsb = (unsigned char*)d_ws;
    size_t off = 0;
    auto carve = [&](size_t bytes) -> void* {
        void* p = wsb + off;
        off = (off + bytes + 255) & ~(size_t)255;
        return p;
    };
    const size_t Nn = N_NODES;

    // All weight buffers hold W^T (shape [N,K] row-major) in bf16.
    bf16_t* zb     = (bf16_t*)carve(Nn * DIN * 2);
    bf16_t* wqkvT  = (bf16_t*)carve((size_t)3 * DIN * DIN * 2);  // [1152,384]
    bf16_t* wprojT = (bf16_t*)carve((size_t)DIN * DIN * 2);      // [384,384]
    bf16_t* wfuseT = (bf16_t*)carve((size_t)EMB * DIN * 2);      // [256,384]
    bf16_t* wt0T   = (bf16_t*)carve((size_t)HD * EMB * 2);       // [512,256]
    bf16_t* wt1T   = (bf16_t*)carve((size_t)HD * DH * 2);        // [512,128]
    bf16_t* ws0T   = (bf16_t*)carve((size_t)DH * DH * 2);        // [128,128]
    bf16_t* ws1T   = (bf16_t*)carve((size_t)DH * DH * 2);        // [128,128]
    float*  qkvf   = (float*) carve(Nn * 3 * DIN * 4);   // later reused for t0,t1
    bf16_t* qbf    = (bf16_t*)carve(Nn * DIN * 2);       // [8192,384]
    bf16_t* kbf    = (bf16_t*)carve(Nn * DIN * 2);       // [8192,384] (= K^T of scores B)
    bf16_t* vT     = (bf16_t*)carve(Nn * DIN * 2);       // [384,8192]
    float*  Sbuf   = (float*) carve((size_t)QBLK * Nn * 4);
    bf16_t* Pbuf   = (bf16_t*)carve((size_t)QBLK * Nn * 2);
    float*  attf   = (float*) carve(Nn * DIN * 4);
    bf16_t* attb   = (bf16_t*)carve(Nn * DIN * 2);
    float*  projf  = (float*) carve(Nn * DIN * 4);
    bf16_t* projbb = (bf16_t*)carve(Nn * DIN * 2);
    float*  fusef  = (float*) carve(Nn * EMB * 4);
    float*  embf   = (float*) carve(Nn * EMB * 4);
    bf16_t* embb   = (bf16_t*)carve(Nn * EMB * 2);
    float*  wia0   = (float*) carve(EMB * NHEAD * 4);
    float*  bia0   = (float*) carve(NHEAD * 4);
    float*  wja0   = (float*) carve(EMB * NHEAD * 4);
    float*  bja0   = (float*) carve(NHEAD * 4);
    float*  wia1   = (float*) carve(DH * NHEAD * 4);
    float*  bia1   = (float*) carve(NHEAD * 4);
    float*  wja1   = (float*) carve(DH * NHEAD * 4);
    float*  bja1   = (float*) carve(NHEAD * 4);
    float*  ali0   = (float*) carve(Nn * NHEAD * 4);
    float*  alj0   = (float*) carve(Nn * NHEAD * 4);
    float*  ali1   = (float*) carve(Nn * NHEAD * 4);
    float*  alj1   = (float*) carve(Nn * NHEAD * 4);
    float*  a0     = (float*) carve((size_t)NEDGE * NHEAD * 4);
    float*  a1     = (float*) carve((size_t)NEDGE * NHEAD * 4);
    float*  out1   = (float*) carve(Nn * DH * 4);
    bf16_t* out1b  = (bf16_t*)carve(Nn * DH * 2);
    float*  out2   = (float*) carve(Nn * DH * 4);
    bf16_t* out2b  = (bf16_t*)carve(Nn * DH * 2);
    float*  s0f    = (float*) carve(Nn * DH * 4);
    float*  s1f    = (float*) carve(Nn * DH * 4);
    // t0/t1 overlay the (dead-after-split) qkvf region: 2*8192*512*4 <= 8192*1152*4
    float* t0 = qkvf;
    float* t1 = qkvf + Nn * HD;

    auto gemm = [&](const bf16_t* A, const bf16_t* Bt, float* C, int M, int Nc,
                    int K, int lda, int ldb, int ldc, const float* bias) {
        dim3 g(Nc / 64, M / 32);
        k_gemm<<<g, 128, 0, stream>>>(A, Bt, C, M, Nc, K, lda, ldb, ldc, bias);
    };
    auto cvt = [&](const float* s, bf16_t* d, long n) {
        k_cvt<<<(int)((n + 255) / 256), 256, 0, stream>>>(s, d, n);
    };
    auto cvtT = [&](const float* s, bf16_t* d, int R, int Cc) {
        k_cvt_t<<<(R * Cc + 255) / 256, 256, 0, stream>>>(s, d, R, Cc);
    };

    // ---- weights -> bf16 transposed ----
    cvtT(qkv_w,  wqkvT,  DIN, 3 * DIN);
    cvtT(proj_w, wprojT, DIN, DIN);
    cvtT(fuse_w, wfuseT, DIN, EMB);
    cvtT(wt0,    wt0T,   EMB, HD);
    cvtT(wt1,    wt1T,   DH,  HD);
    cvtT(skip0_w, ws0T,  DH,  DH);
    cvtT(skip1_w, ws1T,  DH,  DH);

    // ---- z = [tile(x,128) | embed[ids]] ----
    k_prep<<<N_NODES, DIN, 0, stream>>>(ids, x, embed, zb);

    // ---- qkv GEMM + split (q scaled, v transposed) ----
    gemm(zb, wqkvT, qkvf, N_NODES, 3 * DIN, DIN, DIN, DIN, 3 * DIN, qkv_b);
    const float qscale = 1.0f / sqrtf((float)DIN);
    k_split_qkv<<<N_NODES, DIN, 0, stream>>>(qkvf, qbf, kbf, vT, qscale);

    // ---- chunked full attention: softmax(q k^T) v ----
    for (int qb = 0; qb < N_NODES / QBLK; ++qb) {
        const bf16_t* qp = qbf + (size_t)qb * QBLK * DIN;
        // scores: A=q chunk [256,384], Bt=k [8192,384] -> S [256,8192]
        gemm(qp, kbf, Sbuf, QBLK, N_NODES, DIN, DIN, DIN, N_NODES, nullptr);
        k_row_softmax<<<QBLK, 256, 0, stream>>>(Sbuf, Pbuf, N_NODES);
        // P*V: A=P [256,8192], Bt=vT [384,8192] -> att chunk [256,384]
        gemm(Pbuf, vT, attf + (size_t)qb * QBLK * DIN,
             QBLK, DIN, N_NODES, N_NODES, N_NODES, DIN, nullptr);
    }

    // ---- proj, fuse, LN+SiLU+residual ----
    cvt(attf, attb, (long)N_NODES * DIN);
    gemm(attb, wprojT, projf, N_NODES, DIN, DIN, DIN, DIN, DIN, proj_b);
    cvt(projf, projbb, (long)N_NODES * DIN);
    gemm(projbb, wfuseT, fusef, N_NODES, EMB, DIN, DIN, DIN, EMB, fuse_b);
    k_ln_silu_add<<<N_NODES, EMB, 0, stream>>>(fusef, ln1_g, ln1_b, embed, ids, embf, embb);

    // ---- WGAT layers ----
    auto wgat = [&](const float* xin_f, const bf16_t* xin_b, int in_dim,
                    const float* wi, const float* bi, const float* wj, const float* bj,
                    const float* btv, const float* attn, const bf16_t* wtT,
                    float* tbuf, float* wia, float* bia, float* wja, float* bja,
                    float* ali, float* alj, float* abuf, float* outb) {
        const int ft = (in_dim + 1) * NHEAD;
        k_wfold<<<(ft + 127) / 128, 128, 0, stream>>>(wi, bi, attn, wia, bia, in_dim);
        k_wfold<<<(ft + 127) / 128, 128, 0, stream>>>(wj, bj, attn, wja, bja, in_dim);
        k_alpha<<<(N_NODES * NHEAD) / 256, 256, 0, stream>>>(xin_f, wia, bia, ali, in_dim);
        k_alpha<<<(N_NODES * NHEAD) / 256, 256, 0, stream>>>(xin_f, wja, bja, alj, in_dim);
        k_edge_logits<<<NEDGE / 256, 256, 0, stream>>>(ali, alj, src, dst, edge_attr, abuf);
        k_chunk_softmax<<<(NEDGE / CHUNK) * NHEAD, 256, 0, stream>>>(abuf);
        gemm(xin_b, wtT, tbuf, N_NODES, HD, in_dim, in_dim, in_dim, HD, btv);
        k_fill0<<<256, 256, 0, stream>>>(outb, (long)N_NODES * DH);
        k_scatter<<<NEDGE, DH, 0, stream>>>(tbuf, abuf, src, dst, outb);
    };

    wgat(embf, embb, EMB, wi0, bi0, wj0, bj0, bt0, attn0, wt0T, t0,
         wia0, bia0, wja0, bja0, ali0, alj0, a0, out1);
    cvt(out1, out1b, (long)N_NODES * DH);
    gemm(out1b, ws0T, s0f, N_NODES, DH, DH, DH, DH, DH, skip0_b);

    wgat(out1, out1b, DH, wi1, bi1, wj1, bj1, bt1, attn1, wt1T, t1,
         wia1, bia1, wja1, bja1, ali1, alj1, a1, out2);
    cvt(out2, out2b, (long)N_NODES * DH);
    gemm(out2b, ws1T, s1f, N_NODES, DH, DH, DH, DH, DH, skip1_b);

    // ---- final LN + outputs (emb | embed copy | ids) ----
    float* out_emb   = (float*)d_out;
    float* out_embed = out_emb + (size_t)N_NODES * DH;
    float* out_ids   = out_embed + (size_t)50000 * EMB;

    k_final_ln<<<N_NODES, DH, 0, stream>>>(out2, s0f, s1f, ln2_g, ln2_b, out_emb);
    k_copyf<<<2048, 256, 0, stream>>>(embed, out_embed, (long)50000 * EMB);
    k_i2f<<<(N_NODES + 255) / 256, 256, 0, stream>>>(ids, out_ids, N_NODES);
}
